// FocusedLinearCrossAttention_71038759075955
// MI455X (gfx1250) — matrix operand order repaired and checked by reading, not compiled, not run on hardware
//
#include <hip/hip_runtime.h>
#include <hip/hip_bf16.h>
#include <stdint.h>

typedef __attribute__((ext_vector_type(16))) _Float16 v16h;
typedef __attribute__((ext_vector_type(8)))  float    v8f;
typedef __attribute__((ext_vector_type(4)))  unsigned int u32x4;
typedef __attribute__((ext_vector_type(8)))  int      i32x8;
typedef __attribute__((ext_vector_type(4)))  int      i32x4;

#define NB      4
#define NPIX    65536      // pixels per batch (256*256)
#define HWD     256
#define CLMS    32
#define CIN     64         // INNER
#define ROWS    (NB * NPIX)            // 262144 total rows
#define RPB     64                     // rows per block (4 waves * 16)
#define NBLK    (ROWS / RPB)           // 4096 blocks
#define BPB     (NPIX / RPB)           // 1024 blocks per batch
#define PSTRIDE 576                    // per-block partial: 64 ksum + 512 kv2

#ifndef __has_builtin
#define __has_builtin(x) 0
#endif
#if __has_builtin(__builtin_amdgcn_tensor_load_to_lds) && \
    __has_builtin(__builtin_amdgcn_s_wait_tensorcnt)
#define USE_TDM 1
#else
#define USE_TDM 0
#endif

__device__ inline v8f wmma16(v16h a, v16h b, v8f c) {
    // D = A(16x32 f16) * B(32x16 f16) + C(16x16 f32)
    return __builtin_amdgcn_wmma_f32_16x16x32_f16(false, a, false, b,
                                                  (short)0, c, false, false);
}

#if USE_TDM
// Issue a TDM load of a [32 rows x 16 floats] tile (rows strided by NPIX
// floats in global memory) into LDS at byte offset lds_off. D# fields per
// CDNA5 ISA ch.8: group0 = {count|flags, lds_addr, global_addr, type=2},
// group1 = {data_size, tensor dims, tile dims, dim0 stride}.
__device__ inline void tdm_load_tile32x16(const float* gsrc, unsigned lds_off) {
    unsigned long long ga = (unsigned long long)(size_t)gsrc;
    u32x4 g0;
    g0[0] = 1u;                                   // count=1 (valid user D#)
    g0[1] = lds_off;                              // lds_addr (bytes)
    g0[2] = (unsigned)(ga & 0xffffffffu);         // global_addr[31:0]
    g0[3] = (unsigned)((ga >> 32) & 0x01ffffffu)  // global_addr[56:32]
            | (2u << 30);                         // type = 2 ("image")
    const unsigned td0 = NPIX;                    // tensor_dim0 (elements)
    const unsigned td1 = 32;                      // tensor_dim1
    i32x8 g1;
    g1[0] = (int)(2u << 16);                      // data_size = 2 (4 bytes)
    g1[1] = (int)((td0 & 0xffffu) << 16);         // tensor_dim0[15:0] @bit48
    g1[2] = (int)(((td0 >> 16) & 0xffffu) |       // tensor_dim0[31:16]
                  ((td1 & 0xffffu) << 16));       // tensor_dim1[15:0]
    g1[3] = (int)(((td1 >> 16) & 0xffffu) |       // tensor_dim1[31:16]
                  (16u << 16));                   // tile_dim0 = 16
    g1[4] = 32;                                   // tile_dim1 = 32, tile_dim2=0
    g1[5] = (int)NPIX;                            // tensor_dim0_stride[31:0]
    g1[6] = 0;                                    // stride hi + dim1 stride
    g1[7] = 0;
    i32x4 z4 = {0, 0, 0, 0};
#if __clang_major__ >= 23
    i32x8 z8 = {0, 0, 0, 0, 0, 0, 0, 0};
    __builtin_amdgcn_tensor_load_to_lds(g0, g1, z4, z4, z8, 0);
#else
    __builtin_amdgcn_tensor_load_to_lds(g0, g1, z4, z4, 0);
#endif
}
#endif

// ---------------------------------------------------------------------------
// Pass 1: fused QKV GEMM (WMMA, A tiles via TDM) + focusing elementwise +
//         deterministic per-block reduction of ksum / kv2 partials.
// ---------------------------------------------------------------------------
__global__ __launch_bounds__(128) void k_qkv(
    const float* __restrict__ lms, const float* __restrict__ pan,
    const float* __restrict__ qw,  const float* __restrict__ qb,
    const float* __restrict__ kvw, const float* __restrict__ kvb,
    const float* __restrict__ scale,
    float* __restrict__ qf, float* __restrict__ vf, float* __restrict__ part)
{
    __shared__ float sQ[4][16][64];   // also used as TDM A-tile staging
    __shared__ float sK[4][16][64];
    __shared__ float sV[4][16][64];
    __shared__ float sSC[64];

    const int tid = threadIdx.x;
    if (tid < 64) {                          // softplus(scale) once per block
        float s = scale[tid];
        sSC[tid] = logf(1.0f + expf(s));
    }
    const int wave = tid >> 5, lane = tid & 31;
    const int m = lane & 15, hi = lane >> 4;

    const long tile = (long)blockIdx.x * 4 + wave;   // 16-row tile per wave
    const long n0   = tile * 16;
    const int  b    = (int)(n0 >> 16);
    const int  p0   = (int)(n0 & (NPIX - 1));
    const float* lmsb = lms + (size_t)b * CLMS * NPIX;
    const float* panb = pan + (size_t)b * CLMS * NPIX;

    v16h aL, aP;
#if USE_TDM
    // DMA both 16x32 A-tiles into this wave's sQ region (overwritten later
    // by the Q GEMM results, which is safe: in-wave LDS ops stay in order).
    float* stg = &sQ[wave][0][0];                 // 1024 floats per wave
    const unsigned stg_off = (unsigned)(size_t)stg;  // LDS byte offset
    tdm_load_tile32x16(lmsb + p0, stg_off);          // [32][16] lms tile
    tdm_load_tile32x16(panb + p0, stg_off + 2048);   // [32][16] pan tile
    __builtin_amdgcn_s_wait_tensorcnt(0);
    asm volatile("" ::: "memory");                // keep LDS reads below wait
#pragma unroll
    for (int j = 0; j < 16; ++j) {
        int K = hi * 8 + (j < 8 ? j : j + 8);
        aL[j] = (_Float16)stg[K * 16 + m];
        aP[j] = (_Float16)stg[512 + K * 16 + m];
    }
#else
#pragma unroll
    for (int j = 0; j < 16; ++j) {
        int K = hi * 8 + (j < 8 ? j : j + 8);
        aL[j] = (_Float16)lmsb[(size_t)K * NPIX + p0 + m];
        aP[j] = (_Float16)panb[(size_t)K * NPIX + p0 + m];
    }
#endif

    // Q = lms_t @ q_w^T : 4 column tiles of 16
#pragma unroll
    for (int t = 0; t < 4; ++t) {
        const int oc = t * 16 + m;
        v16h bf;
#pragma unroll
        for (int j = 0; j < 16; ++j)
            bf[j] = (_Float16)qw[oc * 32 + hi * 8 + (j < 8 ? j : j + 8)];
        v8f acc = {};
        acc = wmma16(aL, bf, acc);
        const float bias = qb[oc];
#pragma unroll
        for (int r = 0; r < 8; ++r)
            sQ[wave][hi * 8 + r][oc] = acc[r] + bias;
    }

    // KV = pan_t @ kv_w^T : 8 column tiles of 16 (first 64 = K, last 64 = V)
#pragma unroll
    for (int t = 0; t < 8; ++t) {
        const int oc = t * 16 + m;
        v16h bf;
#pragma unroll
        for (int j = 0; j < 16; ++j)
            bf[j] = (_Float16)kvw[oc * 32 + hi * 8 + (j < 8 ? j : j + 8)];
        v8f acc = {};
        acc = wmma16(aP, bf, acc);
        const float bias = kvb[oc];
        if (oc < 64) {
#pragma unroll
            for (int r = 0; r < 8; ++r)
                sK[wave][hi * 8 + r][oc] = acc[r] + bias;
        } else {
#pragma unroll
            for (int r = 0; r < 8; ++r)
                sV[wave][hi * 8 + r][oc - 64] = acc[r] + bias;
        }
    }
    __syncthreads();

    // Focusing: uniform code on all 32 lanes. Lanes 0..15 process the q-row,
    // lanes 16..31 the k-row of the same 16-row tile (pointer select, no
    // divergence). ||x^3|| = sqrt(sum x^6); results written back to LDS.
    {
        const int rl = lane & 15;
        float* srcRow = (lane < 16) ? &sQ[wave][rl][0] : &sK[wave][rl][0];
        float s2 = 0.f, s6 = 0.f;
        for (int c = 0; c < 64; ++c) {
            float e = (fmaxf(srcRow[c], 0.f) + 1e-6f) / sSC[c];
            float e2 = e * e;
            s2 += e2; s6 += e2 * e2 * e2;
        }
        const float ratio = sqrtf(s2 / s6);      // ||x|| / ||x^3||
        for (int c = 0; c < 64; ++c) {
            float e = (fmaxf(srcRow[c], 0.f) + 1e-6f) / sSC[c];
            srcRow[c] = e * e * e * ratio;
        }
    }
    __syncthreads();

    // Coalesced float4 flush of focused q and raw v (LDS layout == global
    // row-major layout for this block's 64 rows).
    {
        const long n0b = (long)blockIdx.x * RPB;
        float4*       qdst = (float4*)(qf + n0b * 64);
        float4*       vdst = (float4*)(vf + n0b * 64);
        const float4* qsrc = (const float4*)&sQ[0][0][0];
        const float4* vsrc = (const float4*)&sV[0][0][0];
        for (int i = tid; i < 1024; i += 128) {
            qdst[i] = qsrc[i];
            vdst[i] = vsrc[i];
        }
    }

    // Deterministic block-local reduction of ksum (64 slots) and kv2 (512).
    float* pblk = part + (size_t)blockIdx.x * PSTRIDE;
    for (int slot = tid; slot < PSTRIDE; slot += 128) {
        float s = 0.f;
        if (slot < 64) {
            const int ch = slot;                   // h*8 + c
            for (int rr = 0; rr < RPB; ++rr)
                s += sK[rr >> 4][rr & 15][ch];
        } else {
            const int s2 = slot - 64;
            const int h = s2 >> 6, c = (s2 >> 3) & 7, d = s2 & 7;
            const int chk = h * 8 + c, chv = h * 8 + d;
            for (int rr = 0; rr < RPB; ++rr)
                s += sK[rr >> 4][rr & 15][chk] * sV[rr >> 4][rr & 15][chv];
        }
        pblk[slot] = s;
    }
}

// ---------------------------------------------------------------------------
// Pass 2: reduce per-block partials -> ksum[32][8], kv2[32][8][8]
// ---------------------------------------------------------------------------
__global__ __launch_bounds__(128) void k_reduce(
    const float* __restrict__ part, float* __restrict__ ksum,
    float* __restrict__ kv2)
{
    const int bh = blockIdx.x;          // 0..31
    const int b = bh >> 3, h = bh & 7;
    const int t = threadIdx.x;
    if (t >= 72) return;
    const float* pb = part + (size_t)b * BPB * PSTRIDE;
    float s = 0.f;
    if (t < 8) {
        for (int i = 0; i < BPB; ++i) s += pb[(size_t)i * PSTRIDE + h * 8 + t];
        ksum[bh * 8 + t] = s;
    } else {
        const int idx = t - 8;          // c*8 + d
        for (int i = 0; i < BPB; ++i)
            s += pb[(size_t)i * PSTRIDE + 64 + h * 64 + idx];
        kv2[bh * 64 + idx] = s;
    }
}

// ---------------------------------------------------------------------------
// Pass 3: x = z*(q@kv2) + depthwise-conv(v) residual, then proj (WMMA),
//         write transposed [b, c, n] output with float4 stores.
//         Each row's 64 channels split across the two lane halves
//         (heads 0-3 vs 4-7) -> all 32 lanes active, branch-free.
// ---------------------------------------------------------------------------
__global__ __launch_bounds__(128) void k_out(
    const float* __restrict__ qf, const float* __restrict__ vf,
    const float* __restrict__ ksum, const float* __restrict__ kv2,
    const float* __restrict__ dwcw, const float* __restrict__ dwcb,
    const float* __restrict__ projw, const float* __restrict__ projb,
    float* __restrict__ out)
{
    __shared__ float sX[4][16][64];
    __shared__ float sKv2[8][8][8];
    __shared__ float sKs[8][8];
    __shared__ float sW[8][25];
    __shared__ float sB2[8];

    const int tid = threadIdx.x;
    const long n0b = (long)blockIdx.x * RPB;
    const int  b   = (int)(n0b >> 16);

    for (int i = tid; i < 512; i += 128)
        ((float*)sKv2)[i] = kv2[(size_t)b * 8 * 64 + i];
    if (tid < 64) ((float*)sKs)[tid] = ksum[b * 64 + tid];
    for (int i = tid; i < 200; i += 128) ((float*)sW)[i] = dwcw[i];
    if (tid < 8) sB2[tid] = dwcb[tid];
    __syncthreads();

    const int wave = tid >> 5, lane = tid & 31;
    const int m = lane & 15, hi = lane >> 4;
    const long n0 = n0b + wave * 16;
    const int  p0 = (int)(n0 & (NPIX - 1));

    {
        const int  rl   = lane & 15;             // row within wave tile
        const int  half = lane >> 4;             // 0: heads 0-3, 1: heads 4-7
        const int  p    = p0 + rl;
        const long row  = (long)b * NPIX + p;
        const int  ay   = p >> 8, ax = p & 255;
        const float* qrow = qf + row * 64 + half * 32;
        __builtin_prefetch(qrow, 0, 1);
        float qv[32];
#pragma unroll
        for (int c = 0; c < 32; ++c) qv[c] = qrow[c];

#pragma unroll
        for (int h2 = 0; h2 < 4; ++h2) {
            const int h = half * 4 + h2;
            float dot = 0.f;
#pragma unroll
            for (int c = 0; c < 8; ++c) dot += qv[h2 * 8 + c] * sKs[h][c];
            const float z = 1.f / (dot + 1e-6f);
#pragma unroll
            for (int d = 0; d < 8; ++d) {
                float s = 0.f;
#pragma unroll
                for (int c = 0; c < 8; ++c)
                    s += qv[h2 * 8 + c] * sKv2[h][c][d];
                const int cc = h * 8 + d;
                // 5x5 depthwise conv on v (channel within head = d)
                float conv = sB2[d];
                for (int dh = -2; dh <= 2; ++dh) {
                    const int yy = ay + dh;
                    if (yy < 0 || yy > 255) continue;
                    for (int dw = -2; dw <= 2; ++dw) {
                        const int xx = ax + dw;
                        if (xx < 0 || xx > 255) continue;
                        conv += sW[d][(dh + 2) * 5 + (dw + 2)] *
                                vf[((size_t)b * NPIX + (yy << 8) + xx) * 64 + cc];
                    }
                }
                sX[wave][rl][cc] = z * s + conv;
            }
        }
    }
    __syncthreads();

    // Projection: out = x @ proj_w^T + proj_b  (K=64 -> two chained WMMAs)
    v16h aF[2];
#pragma unroll
    for (int kk = 0; kk < 2; ++kk)
#pragma unroll
        for (int j = 0; j < 16; ++j) {
            int K = kk * 32 + hi * 8 + (j < 8 ? j : j + 8);
            aF[kk][j] = (_Float16)sX[wave][m][K];
        }
#pragma unroll
    for (int t4 = 0; t4 < 4; ++t4) {
        const int oc = t4 * 16 + m;
        v8f acc = {};
#pragma unroll
        for (int kk = 0; kk < 2; ++kk) {
            v16h bf;
#pragma unroll
            for (int j = 0; j < 16; ++j)
                bf[j] = (_Float16)projw[oc * 64 + kk * 32 + hi * 8 + (j < 8 ? j : j + 8)];
            acc = wmma16(aF[kk], bf, acc);
        }
        const float bias = projb[oc];
        float* obase = out + ((size_t)b * 64 + oc) * NPIX + p0 + hi * 8;
        float4 lo = make_float4(acc[0] + bias, acc[1] + bias,
                                acc[2] + bias, acc[3] + bias);
        float4 hi4 = make_float4(acc[4] + bias, acc[5] + bias,
                                 acc[6] + bias, acc[7] + bias);
        *(float4*)(obase)     = lo;   // 8 consecutive pixels of one channel
        *(float4*)(obase + 4) = hi4;
    }
}

// ---------------------------------------------------------------------------
extern "C" void kernel_launch(void* const* d_in, const int* in_sizes, int n_in,
                              void* d_out, int out_size, void* d_ws, size_t ws_size,
                              hipStream_t stream)
{
    const float* lms   = (const float*)d_in[0];
    const float* pan   = (const float*)d_in[1];
    const float* qw    = (const float*)d_in[2];
    const float* qb    = (const float*)d_in[3];
    const float* kvw   = (const float*)d_in[4];
    const float* kvb   = (const float*)d_in[5];
    const float* projw = (const float*)d_in[6];
    const float* projb = (const float*)d_in[7];
    const float* dwcw  = (const float*)d_in[8];
    const float* dwcb  = (const float*)d_in[9];
    const float* scale = (const float*)d_in[10];

    float* ws   = (float*)d_ws;
    float* qf   = ws;                               // [262144][64]
    float* vf   = qf + (size_t)ROWS * 64;           // [262144][64]
    float* part = vf + (size_t)ROWS * 64;           // [4096][576]
    float* ksum = part + (size_t)NBLK * PSTRIDE;    // [32][8]
    float* kv2  = ksum + 256;                       // [32][64]
    float* outp = (float*)d_out;

    k_qkv<<<NBLK, 128, 0, stream>>>(lms, pan, qw, qb, kvw, kvb, scale,
                                    qf, vf, part);
    k_reduce<<<32, 128, 0, stream>>>(part, ksum, kv2);
    k_out<<<NBLK, 128, 0, stream>>>(qf, vf, ksum, kv2, dwcw, dwcb,
                                    projw, projb, outp);
}
